// MptEigenAttention_80719615361038
// MI455X (gfx1250) — matrix operand order repaired and checked by reading, not compiled, not run on hardware
//
#include <hip/hip_runtime.h>
#include <hip/hip_bf16.h>
#include <math.h>

// ---------------------------------------------------------------------------
// Problem constants (B=1)
// ---------------------------------------------------------------------------
static constexpr int S    = 2048;
static constexpr int D    = 4096;
static constexpr int H    = 32;
static constexpr int DK   = 64;          // rank per head (q/k)
static constexpr int DV   = 64;          // rank per head (v)
static constexpr int RKQ  = 2048;
static constexpr int RV   = 2048;
static constexpr int QKVW = 2 * RKQ + RV;   // 6144 columns of mixed_qkv
static constexpr float SOFTMAX_SCALE = 0.08838834764831845f; // 1/sqrt(128)

typedef __bf16 bf16_t;
typedef __attribute__((ext_vector_type(16))) __bf16 v16bf;
typedef __attribute__((ext_vector_type(8)))  float  v8f;
typedef unsigned int u32;
typedef __attribute__((ext_vector_type(4))) u32 u32x4;
typedef __attribute__((ext_vector_type(8))) int i32x8;
typedef __attribute__((ext_vector_type(4))) int i32x4;

// ---------------------------------------------------------------------------
// WMMA helpers (CDNA5: V_WMMA_F32_16X16X32_BF16, wave32)
// ---------------------------------------------------------------------------
__device__ __forceinline__ v8f wmma_bf16(v16bf a, v16bf b, v8f c) {
  // 8 args: (neg_a, A, neg_b, B, c_mod, C, reuse_a, reuse_b)
  return __builtin_amdgcn_wmma_f32_16x16x32_bf16(
      false, a, false, b, (short)0, c, false, false);
}

// A-fragment loader: 16x32 (MxK) bf16 from row-major [16][ld] memory.
// A layout (matches B loaders): lanes 0-15 (M=lane) hold K=0..15;
// lanes 16-31 (M=lane-16) hold K=16..31.
__device__ __forceinline__ v16bf load_a(const bf16_t* __restrict__ p, int ld) {
  const int lane = threadIdx.x & 31;
  const int row  = lane & 15;
  const int hk   = (lane >> 4) * 16;
  v16bf a;
#pragma unroll
  for (int i = 0; i < 16; ++i) a[i] = p[row * ld + hk + i];
  return a;
}

// B-fragment loader from B^T storage, i.e. memory laid out [N=16 rows][K=32 cols]
// (weights W are [N,K] row-major; B = W^T so element (k,n) = W[n][k]).
// B layout (contiguous k, same order as load_a): lanes 0-15 hold K=0-15 for column N=lane; lanes 16-31 hold K=16-31.
__device__ __forceinline__ v16bf load_bt(const bf16_t* __restrict__ p, int ld) {
  const int lane = threadIdx.x & 31;
  const int n  = lane & 15;
  const int kb = (lane >> 4) * 16;
  v16bf b;
#pragma unroll
  for (int i = 0; i < 16; ++i) b[i] = p[n * ld + kb + i];
  return b;
}

// B-fragment loader from natural B storage, memory [K=32 rows][N=16 cols]
// (used for the V matrix: element (k,n) = V[k][n]).
__device__ __forceinline__ v16bf load_bn(const bf16_t* __restrict__ p, int ld) {
  const int lane = threadIdx.x & 31;
  const int n  = lane & 15;
  const int kb = (lane >> 4) * 16;
  v16bf b;
#pragma unroll
  for (int i = 0; i < 16; ++i) b[i] = p[(kb + i) * ld + n];
  return b;
}

// ---------------------------------------------------------------------------
// TDM: 2D tile DMA global->LDS (bf16 elements). Builds the Tensor DMA
// Descriptor (D#) per CDNA5 ISA and issues
// tensor_load_to_lds. Tracked with TENSORcnt.
//   group0: count=1 | lds_addr | global_addr | type=2
//   group1: data_size=2B, tensor_dim0/1, tile_dim0/1, tensor_dim0_stride
//   group2/3: zero (2D tile: tile_dim2/3 unused, no iterate/gather)
// ---------------------------------------------------------------------------
__device__ __forceinline__ void tdm_load_2d_bf16(
    u32 lds_addr, const void* gptr,
    u32 tensor_d0, u32 tensor_d1, u32 tile_d0, u32 tile_d1, u32 stride_d0) {
  const unsigned long long ga = (unsigned long long)(uintptr_t)gptr;
  u32x4 g0;
  g0.x = 1u;                                          // count=1 valid descriptor
  g0.y = lds_addr;                                    // LDS byte address
  g0.z = (u32)ga;                                     // global_addr[31:0]
  g0.w = (u32)((ga >> 32) & 0x1FFFFFFu) | (2u << 30); // global_addr[56:32] | type=2
  i32x8 g1;
  g1[0] = (int)(1u << 16);                            // data_size=1 (2 bytes)
  g1[1] = (int)((tensor_d0 & 0xFFFFu) << 16);         // tensor_dim0[15:0]
  g1[2] = (int)((tensor_d0 >> 16) | ((tensor_d1 & 0xFFFFu) << 16));
  g1[3] = (int)((tensor_d1 >> 16) | (tile_d0 << 16)); // tile_dim0
  g1[4] = (int)(tile_d1 & 0xFFFFu);                   // tile_dim1, tile_dim2=0
  g1[5] = (int)stride_d0;                             // tensor_dim0_stride[31:0]
  g1[6] = 0;                                          // stride[47:32]=0, dim1_stride=0
  g1[7] = 0;
  const i32x4 z4 = {0, 0, 0, 0};
#if defined(__clang_major__) && (__clang_major__ >= 23)
  const i32x8 z8 = {0, 0, 0, 0, 0, 0, 0, 0};
  __builtin_amdgcn_tensor_load_to_lds(g0, g1, z4, z4, z8, 0);
#else
  __builtin_amdgcn_tensor_load_to_lds(g0, g1, z4, z4, 0);
#endif
}

// ---------------------------------------------------------------------------
// Kernel 1: fp32 -> bf16 cast (grid-stride)
// ---------------------------------------------------------------------------
__global__ void __launch_bounds__(256)
f32_to_bf16(const float* __restrict__ in, bf16_t* __restrict__ out, long n) {
  long i = (long)blockIdx.x * blockDim.x + threadIdx.x;
  const long stride = (long)gridDim.x * blockDim.x;
  for (; i < n; i += stride) out[i] = (bf16_t)in[i];
}

// ---------------------------------------------------------------------------
// Kernel 2/4: C[M,N] = A[M,K] @ B[N,K]^T (+ optional bias), bf16 in, f32 acc.
// Block tile 128x128, BK=32, 256 threads = 8 waves. Tiles staged by the
// Tensor Data Mover (wave 0 issues, s_wait_tensorcnt, workgroup barrier).
// Each wave: rows w*16..w*16+15, all 128 cols -> A-frag reused over 8 WMMAs.
// ---------------------------------------------------------------------------
__global__ void __launch_bounds__(256)
gemm_bf16_wmma(const bf16_t* __restrict__ A, const bf16_t* __restrict__ B,
               int M, int N, int K,
               bf16_t* __restrict__ outBf, float* __restrict__ outF,
               const float* __restrict__ bias) {
  __shared__ bf16_t As[128 * 32];
  __shared__ bf16_t Bs[128 * 32];

  const int tid  = threadIdx.x;
  const int wave = tid >> 5;
  const int lane = tid & 31;
  const int m0   = wave * 16;
  const size_t blockM = (size_t)blockIdx.y * 128;
  const size_t blockN = (size_t)blockIdx.x * 128;

  const u32 ldsA = (u32)(uintptr_t)&As[0];
  const u32 ldsB = (u32)(uintptr_t)&Bs[0];

  const v8f vzero = {};
  v8f acc[8];
#pragma unroll
  for (int g = 0; g < 8; ++g) acc[g] = vzero;

  for (int k0 = 0; k0 < K; k0 += 32) {
    if (wave == 0) {
      // DMA A tile 128x32 and B tile 128x32 into LDS via the TDM.
      tdm_load_2d_bf16(ldsA, A + blockM * (size_t)K + k0,
                       (u32)K, 128u, 32u, 128u, (u32)K);
      tdm_load_2d_bf16(ldsB, B + blockN * (size_t)K + k0,
                       (u32)K, 128u, 32u, 128u, (u32)K);
      __builtin_amdgcn_s_wait_tensorcnt(0);
    }
    __syncthreads();

    const v16bf afrag = load_a(As + m0 * 32, 32);
#pragma unroll
    for (int g = 0; g < 8; ++g) {
      const v16bf bfrag = load_bt(Bs + g * 16 * 32, 32);
      acc[g] = wmma_bf16(afrag, bfrag, acc[g]);
    }
    __syncthreads();
  }

  // C layout: VGPR r -> M = r + (lane>=16 ? 8 : 0); N = lane & 15
  const int n  = lane & 15;
  const int mb = (lane >> 4) * 8;
#pragma unroll
  for (int g = 0; g < 8; ++g) {
#pragma unroll
    for (int r = 0; r < 8; ++r) {
      const size_t row = blockM + m0 + mb + r;
      const size_t col = blockN + g * 16 + n;
      const float v = acc[g][r];
      if (outF) outF[row * N + col] = v + (bias ? bias[col] : 0.0f);
      else      outBf[row * N + col] = (bf16_t)v;
    }
  }
}

// ---------------------------------------------------------------------------
// Kernel 3: causal flash attention with alibi-style bias.
// grid = (S/128, H); 256 threads = 8 waves; wave w handles query rows
// q0 = bx*128 + w*16 .. +15 for head by. Online softmax in fp32.
// qkv is the bf16 mixed projection [S, 6144]: q@0, k@2048, v@4096 (+h*64).
// ---------------------------------------------------------------------------
__global__ void __launch_bounds__(256)
flash_attn_wmma(const bf16_t* __restrict__ qkv,
                const float* __restrict__ posBias,
                bf16_t* __restrict__ ctx) {
  __shared__ bf16_t pS[8 * 16 * 32];   // per-wave 16x32 P tile (C->A relayout)

  const int h    = blockIdx.y;
  const int wave = threadIdx.x >> 5;
  const int lane = threadIdx.x & 31;
  const int q0   = blockIdx.x * 128 + wave * 16;
  const int n    = lane & 15;
  const int mb   = (lane >> 4) * 8;

  // Q fragments (16 rows x 64 rank), loaded once.
  const bf16_t* qbase = qkv + (size_t)q0 * QKVW + h * DK;
  const v16bf qf0 = load_a(qbase, QKVW);        // rank 0..31
  const v16bf qf1 = load_a(qbase + 32, QKVW);   // rank 32..63

  const v8f vzero = {};
  v8f acc[4];
  float m[8], l[8];
#pragma unroll
  for (int g = 0; g < 4; ++g) acc[g] = vzero;
#pragma unroll
  for (int r = 0; r < 8; ++r) { m[r] = -3.0e38f; l[r] = 0.0f; }

  bf16_t* pl = pS + wave * (16 * 32);
  const int kbEnd = (q0 + 47) >> 5;   // 32-key blocks covering keys <= q0+15

  for (int kb = 0; kb < kbEnd; ++kb) {
    const int key0 = kb * 32;

    // Prefetch the next bias rows (lowers to global_prefetch_b8).
    __builtin_prefetch(
        &posBias[((size_t)h * S + q0 + mb) * S + key0 + 32 + n], 0, 1);

    // ---- scores S = Q @ K^T for 32 keys (2 N-groups x 2 K-chunks) ----
    v8f s0 = vzero, s1 = vzero;
    const bf16_t* kbase = qkv + (size_t)key0 * QKVW + RKQ + h * DK;
    {
      const v16bf b00 = load_bt(kbase, QKVW);                       // keys 0-15, rank 0-31
      const v16bf b01 = load_bt(kbase + 32, QKVW);                  // keys 0-15, rank 32-63
      s0 = wmma_bf16(qf0, b00, s0);
      s0 = wmma_bf16(qf1, b01, s0);
      const v16bf b10 = load_bt(kbase + (size_t)16 * QKVW, QKVW);   // keys 16-31
      const v16bf b11 = load_bt(kbase + (size_t)16 * QKVW + 32, QKVW);
      s1 = wmma_bf16(qf0, b10, s1);
      s1 = wmma_bf16(qf1, b11, s1);
    }

    // ---- scale + position bias + causal mask + online softmax (fp32) ----
#pragma unroll
    for (int r = 0; r < 8; ++r) {
      const int qrow = q0 + mb + r;
      const size_t brow = ((size_t)h * S + qrow) * S;
      {
        const int kcol = key0 + n;
        const float v = s0[r] * SOFTMAX_SCALE + posBias[brow + kcol];
        s0[r] = (kcol > qrow) ? -3.0e38f : v;
      }
      {
        const int kcol = key0 + 16 + n;
        const float v = s1[r] * SOFTMAX_SCALE + posBias[brow + kcol];
        s1[r] = (kcol > qrow) ? -3.0e38f : v;
      }
      float rmax = fmaxf(s0[r], s1[r]);
#pragma unroll
      for (int off = 1; off < 16; off <<= 1)
        rmax = fmaxf(rmax, __shfl_xor(rmax, off, 32));
      const float mnew = fmaxf(m[r], rmax);
      const float fold = __expf(m[r] - mnew);
      const float p0 = __expf(s0[r] - mnew);
      const float p1 = __expf(s1[r] - mnew);
      s0[r] = p0;
      s1[r] = p1;
      float rsum = p0 + p1;
#pragma unroll
      for (int off = 1; off < 16; off <<= 1)
        rsum += __shfl_xor(rsum, off, 32);
      l[r] = l[r] * fold + rsum;
      m[r] = mnew;
#pragma unroll
      for (int g = 0; g < 4; ++g) acc[g][r] *= fold;
    }

    // ---- relayout P (C-frag -> row-major LDS -> A-frag) ----
#pragma unroll
    for (int r = 0; r < 8; ++r) {
      pl[(mb + r) * 32 + n]      = (bf16_t)s0[r];
      pl[(mb + r) * 32 + 16 + n] = (bf16_t)s1[r];
    }
    asm volatile("s_wait_dscnt 0" ::: "memory");  // LDS RAW turnaround (same wave)
    const v16bf pfrag = load_a(pl, 32);

    // ---- ctx += P(16x32) @ V(32x64) ----
    const bf16_t* vbase = qkv + (size_t)key0 * QKVW + 2 * RKQ + h * DV;
#pragma unroll
    for (int g = 0; g < 4; ++g) {
      const v16bf bv = load_bn(vbase + g * 16, QKVW);
      acc[g] = wmma_bf16(pfrag, bv, acc[g]);
    }
  }

  // ---- normalize and store context (bf16, merged heads layout [S, 2048]) ----
#pragma unroll
  for (int g = 0; g < 4; ++g) {
#pragma unroll
    for (int r = 0; r < 8; ++r) {
      const int qrow = q0 + mb + r;
      ctx[(size_t)qrow * RV + h * DV + g * 16 + n] =
          (bf16_t)(acc[g][r] / l[r]);
    }
  }
}

// ---------------------------------------------------------------------------
// Host launcher
// ---------------------------------------------------------------------------
extern "C" void kernel_launch(void* const* d_in, const int* in_sizes, int n_in,
                              void* d_out, int out_size, void* d_ws, size_t ws_size,
                              hipStream_t stream) {
  (void)in_sizes; (void)n_in; (void)out_size; (void)ws_size;

  const float* hidden = (const float*)d_in[0];   // [1,2048,4096]
  const float* Wqkv   = (const float*)d_in[1];   // [6144,4096]
  const float* Wout   = (const float*)d_in[2];   // [4096,2048]
  const float* bout   = (const float*)d_in[3];   // [4096]
  const float* pbias  = (const float*)d_in[4];   // [32,2048,2048]
  // d_in[5] = causal mask (triu bool) — reproduced analytically in-kernel.
  float* out = (float*)d_out;                    // [1,2048,4096] fp32

  char* ws = (char*)d_ws;
  size_t off = 0;
  bf16_t* hidden_bf = (bf16_t*)(ws + off); off += (size_t)S * D * 2;        // 16 MB
  bf16_t* Wqkv_bf   = (bf16_t*)(ws + off); off += (size_t)QKVW * D * 2;     // 48 MB
  bf16_t* Wout_bf   = (bf16_t*)(ws + off); off += (size_t)D * RV * 2;       // 16 MB
  bf16_t* qkv_bf    = (bf16_t*)(ws + off); off += (size_t)S * QKVW * 2;     // 24 MB
  bf16_t* ctx_bf    = (bf16_t*)(ws + off); off += (size_t)S * RV * 2;       //  8 MB

  // 1) fp32 -> bf16 casts
  f32_to_bf16<<<2048, 256, 0, stream>>>(hidden, hidden_bf, (long)S * D);
  f32_to_bf16<<<4096, 256, 0, stream>>>(Wqkv, Wqkv_bf, (long)QKVW * D);
  f32_to_bf16<<<2048, 256, 0, stream>>>(Wout, Wout_bf, (long)D * RV);

  // 2) mixed_qkv[S,6144] = hidden[S,4096] @ Wqkv[6144,4096]^T  (bf16 out)
  gemm_bf16_wmma<<<dim3(QKVW / 128, S / 128), 256, 0, stream>>>(
      hidden_bf, Wqkv_bf, S, QKVW, D, qkv_bf, nullptr, nullptr);

  // 3) causal flash attention with position bias -> ctx[S,2048] bf16
  flash_attn_wmma<<<dim3(S / 128, H), 256, 0, stream>>>(qkv_bf, pbias, ctx_bf);

  // 4) out[S,4096] = ctx[S,2048] @ Wout[4096,2048]^T + bout  (fp32 out)
  gemm_bf16_wmma<<<dim3(D / 128, S / 128), 256, 0, stream>>>(
      ctx_bf, Wout_bf, S, D, RV, nullptr, out, bout);
}
